// SharedAttnProc_21028159881594
// MI455X (gfx1250) — matrix operand order repaired and checked by reading, not compiled, not run on hardware
//
#include <hip/hip_runtime.h>

// ---------------------------------------------------------------------------
// Types for CDNA5 WMMA (wave32): v_wmma_f32_16x16x32_bf16
// ---------------------------------------------------------------------------
typedef __attribute__((ext_vector_type(16))) __bf16 v16bf;
typedef __attribute__((ext_vector_type(8)))  __bf16 v8bf;
typedef __attribute__((ext_vector_type(8)))  float  v8f;
typedef __attribute__((ext_vector_type(4)))  unsigned int u32x4;
typedef __attribute__((ext_vector_type(8)))  int i32x8;
typedef __attribute__((ext_vector_type(4)))  int i32x4;

union Frag {        // one WMMA 16-bit A/B operand = 8 VGPRs = two 16B halves
    v16bf v;
    v8bf  h[2];
};

#define NUMV   3
#define HEADS  24
#define HDIM   128
#define CDIM   3072
#define HW     512
#define ENC    512
#define SQ     1024     // enc + hw
#define SK     2048     // enc + hw*NUM

#ifndef __has_builtin
#define __has_builtin(x) 0
#endif
#if __has_builtin(__builtin_amdgcn_tensor_load_to_lds)
#define HAVE_TDM 1
#else
#define HAVE_TDM 0
#endif

static __device__ __forceinline__ v8f vzero8() {
    v8f z = {0.f,0.f,0.f,0.f,0.f,0.f,0.f,0.f};
    return z;
}

#if HAVE_TDM
// Tensor Data Mover: DMA a 32x128 bf16 tile (row-major) from global into LDS,
// padding 4 DWORDs after every 64 DWORDs (one 128-elem row) so the LDS image
// has a 136-element row stride (bank-conflict free for 16-lane row reads).
static __device__ __forceinline__ void tdm_load_tile(unsigned ldsOff,
                                                     unsigned long long ga) {
    u32x4 g0;
    g0[0] = 1u;                                        // count=1, no gather
    g0[1] = ldsOff;                                    // LDS byte address
    g0[2] = (unsigned)(ga & 0xffffffffu);              // global_addr lo
    g0[3] = (unsigned)((ga >> 32) & 0x01ffffffu) | (2u << 30);  // hi | type=2
    i32x8 g1;
    g1[0] = (int)((1u << 16)      // data_size = 1 -> 2 bytes (bf16)
                | (1u << 20)      // pad_enable
                | (5u << 22)      // pad_interval: every 64 DWORDs
                | (3u << 25));    // pad_amount: 4 DWORDs (8 bf16)
    g1[1] = (int)(128u << 16);    // tensor_dim0[15:0]=128 (bits 63:48)
    g1[2] = (int)(32u  << 16);    // tensor_dim0 hi=0 | tensor_dim1[15:0]=32
    g1[3] = (int)(128u << 16);    // tensor_dim1 hi=0 | tile_dim0=128
    g1[4] = 32;                   // tile_dim1=32, tile_dim2=0
    g1[5] = 128;                  // tensor_dim0_stride lo
    g1[6] = 0;                    // stride hi | dim1_stride lo
    g1[7] = 0;
    i32x4 g2 = {0,0,0,0};         // 2D tensor: groups 2/3 unused
    i32x4 g3 = {0,0,0,0};
#if __clang_major__ >= 23
    i32x8 g4 = {0,0,0,0,0,0,0,0};
    __builtin_amdgcn_tensor_load_to_lds(g0, g1, g2, g3, g4, 0);
#else
    __builtin_amdgcn_tensor_load_to_lds(g0, g1, g2, g3, 0);
#endif
}
#endif

// ---------------------------------------------------------------------------
// RoPE tables: Flux 3-axis (16,56,56) pos-embed, repeat_interleave(2) layout.
// ---------------------------------------------------------------------------
__global__ void rope_tables(const int* __restrict__ idsA,
                            const int* __restrict__ idsB,
                            int split, float* __restrict__ cosT,
                            float* __restrict__ sinT) {
    int s = blockIdx.x;
    int d = threadIdx.x;
    const int* row = (s < split) ? (idsA + (size_t)s * 3)
                                 : (idsB + (size_t)(s - split) * 3);
    int axis, off, dim;
    if (d < 16)      { axis = 0; off = 0;  dim = 16; }
    else if (d < 72) { axis = 1; off = 16; dim = 56; }
    else             { axis = 2; off = 72; dim = 56; }
    int p = (d - off) >> 1;
    float freq = powf(10000.0f, -((float)(2 * p)) / (float)dim);
    float f = (float)row[axis] * freq;
    cosT[(size_t)s * HDIM + d] = __cosf(f);
    sinT[(size_t)s * HDIM + d] = __sinf(f);
}

// ---------------------------------------------------------------------------
// Row mapping (roll/concat + attn-out slicing), evaluated once per thread.
// ---------------------------------------------------------------------------
static __device__ __forceinline__ int mapRow(int m, int mode) {
    if (mode == 0) return m;
    if (mode == 1) {
        int Bi = m / (NUMV * HW);
        int rem = m - Bi * (NUMV * HW);
        int i = rem / HW;
        int r = rem - i * HW;
        int src = Bi - i;
        if (src < 0) src += NUMV;
        return src * HW + r;
    }
    if (mode == 2) return (m / HW) * SQ + ENC + (m % HW);
    return (m / HW) * SQ + (m % HW);
}

// ---------------------------------------------------------------------------
// Generic WMMA GEMM:  O[M x 3072] = A'[M x 3072] @ W[3072 x 3072] + bias
// f32 -> bf16 on the fly into double-buffered LDS; bf16 WMMA, f32 accumulate.
// Block 256 threads = 8 waves; tile 128x128, K-step 32; 1 barrier/iteration.
// ---------------------------------------------------------------------------
__global__ __launch_bounds__(256)
void gemm_bf16(const float* __restrict__ A, const float* __restrict__ W,
               const float* __restrict__ bias, float* __restrict__ O,
               int M, int mode) {
    const int lane = threadIdx.x & 31;
    const int wv   = threadIdx.x >> 5;
    const int wr   = wv & 3;          // wave row (0..3)  -> 32 rows each
    const int wc   = wv >> 2;         // wave col (0..1)  -> 64 cols each
    const int nn   = lane & 15;
    const int half = lane >> 4;
    const int nBase = blockIdx.x * 128;
    const int mBase = blockIdx.y * 128;

    __shared__ __bf16 sA[2][128][40];   // [buf][m][k], 80B row stride
    __shared__ __bf16 sW[2][128][40];   // [buf][n][k] (transposed weights)

    // Hoisted per-thread staging coordinates + source pointers (mode branch
    // resolved once, not in the hot loop).
    const float* aPtr[4];
    const float* wPtr[4];
    int aRow[4], aKq[4], wKk[4], wNq[4];
    for (int i = 0; i < 4; ++i) {
        int g = threadIdx.x + 256 * i;
        aRow[i] = g >> 3;
        aKq[i]  = (g & 7) * 4;
        aPtr[i] = A + (size_t)mapRow(mBase + aRow[i], mode) * CDIM + aKq[i];
        wKk[i]  = g >> 5;
        wNq[i]  = (g & 31) * 4;
        wPtr[i] = W + (size_t)wKk[i] * CDIM + nBase + wNq[i];
    }

    v8f acc[2][4];
    for (int mt = 0; mt < 2; ++mt)
        for (int nt = 0; nt < 4; ++nt) acc[mt][nt] = vzero8();

    // ---- staging helper (k0 slice -> LDS buffer b) ----
    auto stage = [&](int k0, int b) {
        for (int i = 0; i < 4; ++i) {
            float4 av = *(const float4*)(aPtr[i] + k0);
            sA[b][aRow[i]][aKq[i] + 0] = (__bf16)av.x;
            sA[b][aRow[i]][aKq[i] + 1] = (__bf16)av.y;
            sA[b][aRow[i]][aKq[i] + 2] = (__bf16)av.z;
            sA[b][aRow[i]][aKq[i] + 3] = (__bf16)av.w;
            float4 wv4 = *(const float4*)(wPtr[i] + (size_t)k0 * CDIM);
            sW[b][wNq[i] + 0][wKk[i]] = (__bf16)wv4.x;
            sW[b][wNq[i] + 1][wKk[i]] = (__bf16)wv4.y;
            sW[b][wNq[i] + 2][wKk[i]] = (__bf16)wv4.z;
            sW[b][wNq[i] + 3][wKk[i]] = (__bf16)wv4.w;
            if (k0 + 64 < CDIM) {
                __builtin_prefetch(aPtr[i] + k0 + 64, 0, 1);
                __builtin_prefetch(wPtr[i] + (size_t)(k0 + 64) * CDIM, 0, 1);
            }
        }
    };

    stage(0, 0);
    __syncthreads();

    const int ko = half ? 8 : 0;     // A layout: lanes>=16 start at K=8/K=24
    const int kb = half ? 8 : 0;     // B layout: lanes>=16 start at K=8/K=24
    int buf = 0;
    for (int k0 = 0; k0 < CDIM; k0 += 32) {
        if (k0 + 32 < CDIM) stage(k0 + 32, buf ^ 1);

        Frag af[2];
        for (int mt = 0; mt < 2; ++mt) {
            int row = 32 * wr + 16 * mt + nn;
            af[mt].h[0] = *(const v8bf*)&sA[buf][row][ko];
            af[mt].h[1] = *(const v8bf*)&sA[buf][row][ko + 16];
        }
        for (int nt = 0; nt < 4; ++nt) {
            Frag bf_;
            int col = 64 * wc + 16 * nt + nn;
            bf_.h[0] = *(const v8bf*)&sW[buf][col][kb];
            bf_.h[1] = *(const v8bf*)&sW[buf][col][kb + 16];
            for (int mt = 0; mt < 2; ++mt)
                acc[mt][nt] = __builtin_amdgcn_wmma_f32_16x16x32_bf16(
                    false, af[mt].v, false, bf_.v, (short)0, acc[mt][nt],
                    false, false);
        }
        __syncthreads();
        buf ^= 1;
    }

    // ---- epilogue: C layout row = g + 8*half, col = nn ----
    for (int mt = 0; mt < 2; ++mt)
        for (int nt = 0; nt < 4; ++nt) {
            int gcol = nBase + 64 * wc + 16 * nt + nn;
            float b = bias[gcol];
            for (int g = 0; g < 8; ++g) {
                int grow = mBase + 32 * wr + 16 * mt + g + 8 * half;
                O[(size_t)grow * CDIM + gcol] = acc[mt][nt][g] + b;
            }
        }
}

// ---------------------------------------------------------------------------
// Pack Q/K/V: rms_norm (flag 1) + rotary (flag 2) + f32 -> bf16 into
// (B, H, S, 128).  grid = (S, HEADS, B), block = 128.
// ---------------------------------------------------------------------------
__global__ void pack_qkv(const float* __restrict__ encP,
                         const float* __restrict__ imgP,
                         const float* __restrict__ encW,
                         const float* __restrict__ imgW,
                         const float* __restrict__ cosT,
                         const float* __restrict__ sinT,
                         __bf16* __restrict__ out,
                         int S, int imgRowsPerB, int flags) {
    int s = blockIdx.x, h = blockIdx.y, bi = blockIdx.z, d = threadIdx.x;
    const float* src;
    const float* w;
    if (s < ENC) {
        src = encP + ((size_t)(bi * ENC + s)) * CDIM + h * HDIM;
        w = encW;
    } else {
        src = imgP + ((size_t)(bi * imgRowsPerB + (s - ENC))) * CDIM + h * HDIM;
        w = imgW;
    }
    float x = src[d];

    float ss = x * x;
    for (int o = 16; o; o >>= 1) ss += __shfl_xor(ss, o, 32);
    __shared__ float ws4[4];
    if ((threadIdx.x & 31) == 0) ws4[threadIdx.x >> 5] = ss;
    __syncthreads();
    float tot = ws4[0] + ws4[1] + ws4[2] + ws4[3];

    float xn = x;
    if (flags & 1) xn = x * rsqrtf(tot * (1.0f / HDIM) + 1e-6f) * w[d];
    float y = xn;
    if (flags & 2) {
        float xr  = __shfl_xor(xn, 1, 32);              // paired element
        float rot = (d & 1) ? xr : -xr;                  // [-x1, x0] pattern
        y = xn * cosT[(size_t)s * HDIM + d] + rot * sinT[(size_t)s * HDIM + d];
    }
    out[(((size_t)(bi * HEADS + h)) * S + s) * HDIM + d] = (__bf16)y;
}

// ---------------------------------------------------------------------------
// Flash attention.  grid = (SQ/128, HEADS, B), block = 256 (8 waves).
// K chunks staged by the Tensor Data Mover (wave 0 issues, s_wait_tensorcnt);
// V staged transposed by all threads.  Online softmax in WMMA C layout.
// ---------------------------------------------------------------------------
__global__ __launch_bounds__(256)
void attn_fa(const __bf16* __restrict__ Q, const __bf16* __restrict__ K,
             const __bf16* __restrict__ V, const float* __restrict__ mask,
             const int* __restrict__ tsPtr, float* __restrict__ Oout) {
    const int lane = threadIdx.x & 31;
    const int wv   = threadIdx.x >> 5;
    const int nn   = lane & 15;
    const int half = lane >> 4;
    const int h  = blockIdx.y;
    const int bi = blockIdx.z;
    const int qRowBase = blockIdx.x * 128 + wv * 16;
    const bool useMask = (tsPtr[0] < 1);
    const float scale = 0.08838834764831845f;   // 128^-0.5

    __shared__ __bf16 lK[32][136];       // K chunk, 136-elem stride (TDM pad)
    __shared__ __bf16 lVt[128][40];      // V chunk transposed: [n][k]
    __shared__ __bf16 lP[8][16][40];     // per-wave P staging (C->A relayout)

    // ---- Q fragments (16 x 128 = 4 A-frags), direct from global ----
    Frag qf[4];
    {
        const __bf16* Qp =
            Q + (((size_t)(bi * HEADS + h)) * SQ + qRowBase + nn) * HDIM;
        const int ko = half ? 8 : 0;
        for (int c = 0; c < 4; ++c) {
            qf[c].h[0] = *(const v8bf*)(Qp + 32 * c + ko);
            qf[c].h[1] = *(const v8bf*)(Qp + 32 * c + ko + 16);
        }
    }

    v8f oacc[8];
    for (int i = 0; i < 8; ++i) oacc[i] = vzero8();
    float mrow[8], lrow[8];
    for (int g = 0; g < 8; ++g) { mrow[g] = -1e30f; lrow[g] = 0.f; }

    const __bf16* Kb = K + ((size_t)(bi * HEADS + h)) * SK * HDIM;
    const __bf16* Vb = V + ((size_t)(bi * HEADS + h)) * SK * HDIM;
    const int kb = half ? 8 : 0;
#if HAVE_TDM
    const unsigned ldsK = (unsigned)(uintptr_t)&lK[0][0];
#endif

    for (int j = 0; j < SK; j += 32) {
#if HAVE_TDM
        // ---- K chunk via Tensor Data Mover (one DMA per workgroup) ----
        if (wv == 0)
            tdm_load_tile(ldsK,
                          (unsigned long long)(uintptr_t)Kb +
                              (unsigned long long)j * HDIM * 2);
#endif
        // ---- V chunk staged transposed by all threads ----
        for (int t = 0; t < 2; ++t) {
            int g   = threadIdx.x + 256 * t;
            int row = g >> 4;
            int nb  = (g & 15) * 8;
#if !HAVE_TDM
            v8bf kv = *(const v8bf*)(Kb + (size_t)(j + row) * HDIM + nb);
            *(v8bf*)&lK[row][nb] = kv;
#endif
            v8bf vv = *(const v8bf*)(Vb + (size_t)(j + row) * HDIM + nb);
            for (int e = 0; e < 8; ++e) lVt[nb + e][row] = vv[e];
        }
#if HAVE_TDM
        if (wv == 0) __builtin_amdgcn_s_wait_tensorcnt(0);
#endif
        __syncthreads();

        // ---- S = Q K^T : two 16x16 tiles over the 32 K-rows ----
        v8f s0 = vzero8(), s1 = vzero8();
        for (int c = 0; c < 4; ++c) {
            Frag b0, b1;
            b0.h[0] = *(const v8bf*)&lK[nn][32 * c + kb];
            b0.h[1] = *(const v8bf*)&lK[nn][32 * c + kb + 16];
            b1.h[0] = *(const v8bf*)&lK[16 + nn][32 * c + kb];
            b1.h[1] = *(const v8bf*)&lK[16 + nn][32 * c + kb + 16];
            s0 = __builtin_amdgcn_wmma_f32_16x16x32_bf16(
                false, qf[c].v, false, b0.v, (short)0, s0, false, false);
            s1 = __builtin_amdgcn_wmma_f32_16x16x32_bf16(
                false, qf[c].v, false, b1.v, (short)0, s1, false, false);
        }

        // ---- scale + mask + online softmax (per-row, C layout) ----
        float p0[8], p1[8], alpha[8];
        for (int g = 0; g < 8; ++g) {
            float v0 = s0[g] * scale;
            float v1 = s1[g] * scale;
            if (useMask) {
                int qr = qRowBase + g + 8 * half;
                v0 += mask[(size_t)qr * SK + j + nn];
                v1 += mask[(size_t)qr * SK + j + 16 + nn];
            }
            float rm = fmaxf(v0, v1);
            for (int o = 1; o < 16; o <<= 1) rm = fmaxf(rm, __shfl_xor(rm, o, 32));
            float mnew = fmaxf(mrow[g], rm);
            float al = __expf(mrow[g] - mnew);
            float e0 = __expf(v0 - mnew);
            float e1 = __expf(v1 - mnew);
            float rs = e0 + e1;
            for (int o = 1; o < 16; o <<= 1) rs += __shfl_xor(rs, o, 32);
            lrow[g] = lrow[g] * al + rs;
            mrow[g] = mnew;
            alpha[g] = al; p0[g] = e0; p1[g] = e1;
        }
        for (int nt = 0; nt < 8; ++nt)
            for (int g = 0; g < 8; ++g) oacc[nt][g] *= alpha[g];

        // ---- C-layout P -> A-layout via per-wave LDS ----
        for (int g = 0; g < 8; ++g) {
            lP[wv][g + 8 * half][nn]      = (__bf16)p0[g];
            lP[wv][g + 8 * half][16 + nn] = (__bf16)p1[g];
        }
        asm volatile("s_wait_dscnt 0" ::: "memory");
        Frag pf;
        {
            const int ko = half ? 8 : 0;
            pf.h[0] = *(const v8bf*)&lP[wv][nn][ko];
            pf.h[1] = *(const v8bf*)&lP[wv][nn][ko + 16];
        }

        // ---- O += P @ V : 8 col-tiles of 16 over HDIM=128 ----
        for (int nt = 0; nt < 8; ++nt) {
            Frag bv_;
            int col = nt * 16 + nn;
            bv_.h[0] = *(const v8bf*)&lVt[col][kb];
            bv_.h[1] = *(const v8bf*)&lVt[col][kb + 16];
            oacc[nt] = __builtin_amdgcn_wmma_f32_16x16x32_bf16(
                false, pf.v, false, bv_.v, (short)0, oacc[nt], false, false);
        }
        __syncthreads();
    }

    // ---- epilogue: O / l  -> (B, SQ, 3072) with heads concatenated ----
    float inv[8];
    for (int g = 0; g < 8; ++g) inv[g] = 1.0f / lrow[g];
    float* Op = Oout + (size_t)bi * SQ * CDIM + h * HDIM;
    for (int nt = 0; nt < 8; ++nt)
        for (int g = 0; g < 8; ++g) {
            int row = qRowBase + g + 8 * half;
            Op[(size_t)row * CDIM + nt * 16 + nn] = oacc[nt][g] * inv[g];
        }
}

// ---------------------------------------------------------------------------
// Host-side launch
// ---------------------------------------------------------------------------
extern "C" void kernel_launch(void* const* d_in, const int* in_sizes, int n_in,
                              void* d_out, int out_size, void* d_ws,
                              size_t ws_size, hipStream_t stream) {
    (void)in_sizes; (void)n_in; (void)out_size; (void)ws_size;

    const float* hs   = (const float*)d_in[0];
    const float* ehs  = (const float*)d_in[1];
    const float* msk  = (const float*)d_in[2];
    const float* Wq   = (const float*)d_in[3];
    const float* bq   = (const float*)d_in[4];
    const float* Wk   = (const float*)d_in[5];
    const float* bk   = (const float*)d_in[6];
    const float* Wv   = (const float*)d_in[7];
    const float* bv   = (const float*)d_in[8];
    const float* Waq  = (const float*)d_in[9];
    const float* baq  = (const float*)d_in[10];
    const float* Wak  = (const float*)d_in[11];
    const float* bak  = (const float*)d_in[12];
    const float* Wav  = (const float*)d_in[13];
    const float* bav  = (const float*)d_in[14];
    const float* Wo   = (const float*)d_in[15];
    const float* bo   = (const float*)d_in[16];
    const float* Wao  = (const float*)d_in[17];
    const float* bao  = (const float*)d_in[18];
    const float* nq_w  = (const float*)d_in[19];
    const float* nk_w  = (const float*)d_in[20];
    const float* naq_w = (const float*)d_in[21];
    const float* nak_w = (const float*)d_in[22];
    const int* txt_ids = (const int*)d_in[23];
    const int* img_ids = (const int*)d_in[24];
    const int* img_ids_cat = (const int*)d_in[25];
    const int* tstep   = (const int*)d_in[26];

    char* ws = (char*)d_ws;
    size_t off = 0;
    auto alloc = [&](size_t bytes) {
        size_t r = off;
        off += (bytes + 255) & ~(size_t)255;
        return r;
    };
    float*  encS = (float*)(ws + alloc((size_t)1536 * CDIM * 4));
    float*  imgS = (float*)(ws + alloc((size_t)4608 * CDIM * 4));
    __bf16* Qb   = (__bf16*)(ws + alloc((size_t)3 * HEADS * SQ * HDIM * 2));
    __bf16* Kb   = (__bf16*)(ws + alloc((size_t)3 * HEADS * SK * HDIM * 2));
    __bf16* Vb   = (__bf16*)(ws + alloc((size_t)3 * HEADS * SK * HDIM * 2));
    float* cosq  = (float*)(ws + alloc((size_t)SQ * HDIM * 4));
    float* sinq  = (float*)(ws + alloc((size_t)SQ * HDIM * 4));
    float* cosk  = (float*)(ws + alloc((size_t)SK * HDIM * 4));
    float* sink  = (float*)(ws + alloc((size_t)SK * HDIM * 4));
    float* attnO = imgS;   // reuse: img-projection scratch is dead by then

    rope_tables<<<SQ, 128, 0, stream>>>(txt_ids, img_ids, ENC, cosq, sinq);
    rope_tables<<<SK, 128, 0, stream>>>(txt_ids, img_ids_cat, ENC, cosk, sink);

    dim3 blk(256);
    // ---- Q ----
    gemm_bf16<<<dim3(24, 12), blk, 0, stream>>>(ehs, Waq, baq, encS, 1536, 0);
    gemm_bf16<<<dim3(24, 12), blk, 0, stream>>>(hs,  Wq,  bq,  imgS, 1536, 0);
    pack_qkv<<<dim3(SQ, HEADS, 3), 128, 0, stream>>>(
        encS, imgS, naq_w, nq_w, cosq, sinq, Qb, SQ, HW, 3);
    // ---- K (multi-view roll/concat via mode 1) ----
    gemm_bf16<<<dim3(24, 12), blk, 0, stream>>>(ehs, Wak, bak, encS, 1536, 0);
    gemm_bf16<<<dim3(24, 36), blk, 0, stream>>>(hs,  Wk,  bk,  imgS, 4608, 1);
    pack_qkv<<<dim3(SK, HEADS, 3), 128, 0, stream>>>(
        encS, imgS, nak_w, nk_w, cosk, sink, Kb, SK, NUMV * HW, 3);
    // ---- V ----
    gemm_bf16<<<dim3(24, 12), blk, 0, stream>>>(ehs, Wav, bav, encS, 1536, 0);
    gemm_bf16<<<dim3(24, 36), blk, 0, stream>>>(hs,  Wv,  bv,  imgS, 4608, 1);
    pack_qkv<<<dim3(SK, HEADS, 3), 128, 0, stream>>>(
        encS, imgS, nullptr, nullptr, nullptr, nullptr, Vb, SK, NUMV * HW, 0);
    // ---- attention ----
    attn_fa<<<dim3(SQ / 128, HEADS, 3), blk, 0, stream>>>(
        Qb, Kb, Vb, msk, tstep, attnO);
    // ---- output projections (tuple order: img_out then enc_out) ----
    float* outImg = (float*)d_out;
    float* outEnc = outImg + (size_t)1536 * CDIM;
    gemm_bf16<<<dim3(24, 12), blk, 0, stream>>>(attnO, Wo,  bo,  outImg, 1536, 2);
    gemm_bf16<<<dim3(24, 12), blk, 0, stream>>>(attnO, Wao, bao, outEnc, 1536, 3);
}
